// SAGPooling_13589276525190
// MI455X (gfx1250) — matrix operand order repaired and checked by reading, not compiled, not run on hardware
//
#include <hip/hip_runtime.h>
#include <hip/hip_bf16.h>
#include <math.h>

// ---------------- problem constants (static shapes from the reference) ------
#define BG   32            // graphs
#define NN   2048          // nodes per graph
#define NTOT (BG * NN)     // 65536
#define EPER 16384
#define ET   (BG * EPER)   // 524288 edges
#define DD   128           // channels
#define KK   1024          // kept nodes per graph

typedef __attribute__((ext_vector_type(16))) _Float16 v16h;
typedef __attribute__((ext_vector_type(8)))  float    v8f;

// ---------------------------------------------------------------------------
// K0: init scratch: s[] = 0, new_idx[] = -1
// ---------------------------------------------------------------------------
__global__ void k_init(float* __restrict__ s, int* __restrict__ new_idx) {
    int i = blockIdx.x * blockDim.x + threadIdx.x;
    if (i < NTOT) { s[i] = 0.0f; new_idx[i] = -1; }
}

// ---------------------------------------------------------------------------
// K1: WMMA dual-GEMV. Per wave: one 16-node tile.
//   A (16x32 f16 per chunk) = x rows; B (32x16 f16) columns 0-7 = w_rel,
//   columns 8-15 = w_root. D[m][col] = (col<8 ? p : q) of node m.
//   4 chained v_wmma_f32_16x16x32_f16 cover D=128.
// ---------------------------------------------------------------------------
__global__ __launch_bounds__(256)
void k_score_dots(const float* __restrict__ x,
                  const float* __restrict__ w_rel,
                  const float* __restrict__ w_root,
                  float* __restrict__ p, float* __restrict__ q) {
    const int wave = (blockIdx.x * blockDim.x + threadIdx.x) >> 5;
    const int lane = threadIdx.x & 31;
    const int base = wave * 16;              // grid sized exactly: base < NTOT
    const int hi   = lane >> 4;              // half-wave select
    const int m    = lane & 15;              // row (A) / column (B,D)

    // B fragments: lane holds column m; K rows match A layout
    // (lanes 0-15 -> K = 0..7 and 16..23 of the chunk, lanes 16-31: +8).
    const float* wsel = (m < 8) ? w_rel : w_root;
    v16h bfrag[4];
#pragma unroll
    for (int c = 0; c < 4; ++c) {
#pragma unroll
        for (int h = 0; h < 16; ++h)
            bfrag[c][h] = (_Float16)wsel[32 * c + hi * 8 + (h & 7) + ((h >> 3) << 4)];
    }

    // A fragments: lane holds row (base+m); ISA 16-bit A layout:
    // lanes 0-15: K = kb..kb+7 and kb+16..kb+23 (kb = 32c), lanes 16-31: +8.
    const float* xrow = x + (size_t)(base + m) * DD;
    v8f acc = {};
#pragma unroll
    for (int c = 0; c < 4; ++c) {
        const int kb = 32 * c + hi * 8;
        const float4 f0 = *(const float4*)(xrow + kb);
        const float4 f1 = *(const float4*)(xrow + kb + 4);
        const float4 f2 = *(const float4*)(xrow + kb + 16);
        const float4 f3 = *(const float4*)(xrow + kb + 20);
        v16h a;
        a[0]  = (_Float16)f0.x; a[1]  = (_Float16)f0.y;
        a[2]  = (_Float16)f0.z; a[3]  = (_Float16)f0.w;
        a[4]  = (_Float16)f1.x; a[5]  = (_Float16)f1.y;
        a[6]  = (_Float16)f1.z; a[7]  = (_Float16)f1.w;
        a[8]  = (_Float16)f2.x; a[9]  = (_Float16)f2.y;
        a[10] = (_Float16)f2.z; a[11] = (_Float16)f2.w;
        a[12] = (_Float16)f3.x; a[13] = (_Float16)f3.y;
        a[14] = (_Float16)f3.z; a[15] = (_Float16)f3.w;
        acc = __builtin_amdgcn_wmma_f32_16x16x32_f16(
            /*neg_a=*/false, a, /*neg_b=*/false, bfrag[c],
            /*c_mod=*/(short)0, acc, /*reuse_a=*/false, /*reuse_b=*/false);
    }

    // D layout: VGPR r, lanes 0-15 -> M=r col=lane; lanes 16-31 -> M=8+r.
    // Lane picks VGPR (lane&7): all 32 lanes emit distinct (node, p|q) scalars.
    float val = acc[0];
#pragma unroll
    for (int r = 1; r < 8; ++r)
        if ((lane & 7) == r) val = acc[r];
    float* dst = (m < 8) ? p : q;
    dst[base + hi * 8 + (lane & 7)] = val;
}

// ---------------------------------------------------------------------------
// K2: scalar edge aggregation: s[dst] += p[src]   (4 B/edge instead of 512 B)
// ---------------------------------------------------------------------------
__global__ void k_edge_agg(const int* __restrict__ ei,
                           const float* __restrict__ p,
                           float* __restrict__ s) {
    int e = blockIdx.x * blockDim.x + threadIdx.x;
    if (e < ET) {
        int srcn = ei[e];
        int dstn = ei[ET + e];
        atomicAdd(&s[dstn], p[srcn]);
    }
}

// ---------------------------------------------------------------------------
// K3: per-graph score + bitonic top-K (descending, stable like lax.top_k).
//     One 1024-thread block per graph; 2048 (key,idx) pairs in 16 KB LDS.
// ---------------------------------------------------------------------------
__global__ __launch_bounds__(1024)
void k_topk(const float* __restrict__ s, const float* __restrict__ q,
            const float* __restrict__ bias,
            float* __restrict__ o_batch, float* __restrict__ o_perm,
            float* __restrict__ o_score, int* __restrict__ new_idx) {
    __shared__ float key[NN];
    __shared__ int   kid[NN];
    const int g = blockIdx.x;
    const int t = threadIdx.x;
    const float b0 = bias[0];

    for (int i = t; i < NN; i += 1024) {
        int node = g * NN + i;
        key[i] = tanhf(s[node] + b0 + q[node]);
        kid[i] = i;
    }
    __syncthreads();

    // bitonic sort, final order: descending by key, ascending idx on ties
    for (int k = 2; k <= NN; k <<= 1) {
        for (int j = k >> 1; j > 0; j >>= 1) {
            int i   = ((t & ~(j - 1)) << 1) | (t & (j - 1));
            int ixj = i + j;
            float ka = key[i], kb = key[ixj];
            int   ia = kid[i], ib = kid[ixj];
            bool descBlock = ((i & k) == 0);
            // before(a,b): a precedes b in descending stable order
            bool bBeforeA = (kb > ka) || (kb == ka && ib < ia);
            bool doSwap = descBlock ? bBeforeA : !bBeforeA;
            if (doSwap) { key[i] = kb; key[ixj] = ka; kid[i] = ib; kid[ixj] = ia; }
            __syncthreads();
        }
    }

    // first K entries are the (sorted) kept nodes
    int li    = kid[t];
    float sv  = key[t];
    int gnode = g * NN + li;
    int orow  = g * KK + t;
    o_perm[orow]  = (float)gnode;
    o_score[orow] = sv;
    o_batch[orow] = (float)g;
    new_idx[gnode] = orow;
}

// ---------------------------------------------------------------------------
// K4: x_out[row] = x[perm[row]] * score[row]; one wave per 128-float row,
//     float4 per lane -> fully coalesced 512 B per wave.
// ---------------------------------------------------------------------------
__global__ __launch_bounds__(256)
void k_gather_rows(const float* __restrict__ x,
                   const float* __restrict__ perm_f,
                   const float* __restrict__ score_f,
                   float* __restrict__ x_out) {
    const int row  = (blockIdx.x * blockDim.x + threadIdx.x) >> 5;
    const int lane = threadIdx.x & 31;
    if (row >= BG * KK) return;
    const int node = (int)(perm_f[row] + 0.5f);   // exact for values < 2^24
    const float sc = score_f[row];
    const float4* src = (const float4*)(x + (size_t)node * DD);
    float4*       dst = (float4*)(x_out + (size_t)row * DD);
    float4 v = src[lane];
    v.x *= sc; v.y *= sc; v.z *= sc; v.w *= sc;
    dst[lane] = v;
}

// ---------------------------------------------------------------------------
// K5: edge re-indexing + mask
// ---------------------------------------------------------------------------
__global__ void k_edge_out(const int* __restrict__ ei,
                           const int* __restrict__ new_idx,
                           float* __restrict__ o_ei,
                           float* __restrict__ o_emask) {
    int e = blockIdx.x * blockDim.x + threadIdx.x;
    if (e < ET) {
        int r = new_idx[ei[e]];
        int c = new_idx[ei[ET + e]];
        bool em = (r >= 0) && (c >= 0);
        o_ei[e]        = em ? (float)r : -1.0f;
        o_ei[ET + e]   = em ? (float)c : -1.0f;
        o_emask[e]     = em ? 1.0f : 0.0f;
    }
}

// ---------------------------------------------------------------------------
extern "C" void kernel_launch(void* const* d_in, const int* in_sizes, int n_in,
                              void* d_out, int out_size, void* d_ws, size_t ws_size,
                              hipStream_t stream) {
    (void)in_sizes; (void)n_in; (void)out_size; (void)ws_size;

    const float* x      = (const float*)d_in[0];
    const int*   ei     = (const int*)  d_in[1];
    // d_in[2] = batch (recomputed as graph id, unused)
    const float* w_root = (const float*)d_in[3];
    const float* w_rel  = (const float*)d_in[4];
    const float* bias   = (const float*)d_in[5];

    // output layout: x_out | edge_index_out | batch_out | perm | score | emask
    float* out     = (float*)d_out;
    float* o_x     = out;
    float* o_ei    = o_x     + (size_t)BG * KK * DD;
    float* o_batch = o_ei    + (size_t)2 * ET;
    float* o_perm  = o_batch + (size_t)BG * KK;
    float* o_score = o_perm  + (size_t)BG * KK;
    float* o_emask = o_score + (size_t)BG * KK;

    // scratch: p, q, s (floats) + new_idx (int) = 1 MB
    float* p       = (float*)d_ws;
    float* q       = p + NTOT;
    float* s       = q + NTOT;
    int*   new_idx = (int*)(s + NTOT);

    // K0: init scratch
    k_init<<<(NTOT + 255) / 256, 256, 0, stream>>>(s, new_idx);

    // K1: p = x@w_rel, q = x@w_root via WMMA (4096 tiles, 8 waves/block)
    k_score_dots<<<(NTOT / 16) / 8, 256, 0, stream>>>(x, w_rel, w_root, p, q);

    // K2: s[dst] += p[src]
    k_edge_agg<<<(ET + 255) / 256, 256, 0, stream>>>(ei, p, s);

    // K3: per-graph tanh + top-K sort; writes perm/score/batch/new_idx
    k_topk<<<BG, 1024, 0, stream>>>(s, q, bias, o_batch, o_perm, o_score, new_idx);

    // K4: x_out gather+scale (wave per row)
    k_gather_rows<<<(BG * KK) / 8, 256, 0, stream>>>(x, o_perm, o_score, o_x);

    // K5: edge re-index + mask
    k_edge_out<<<(ET + 255) / 256, 256, 0, stream>>>(ei, new_idx, o_ei, o_emask);
}
